// LocalModel_15075335209144
// MI455X (gfx1250) — matrix operand order repaired and checked by reading, not compiled, not run on hardware
//
#include <hip/hip_runtime.h>
#include <hip/hip_bf16.h>
#include <math.h>

#define N_NODES 12000
#define N_EDGES 384000
#define F_IN    512
#define HDIM    64
#define NCLS    7
#define KW      6        // waves (= i-tiles) per block in the kappa kernel
#define LDS_ROW 40       // halfs per padded LDS row (80 bytes)

typedef _Float16     v16h  __attribute__((ext_vector_type(16)));
typedef _Float16     half8 __attribute__((ext_vector_type(8)));
typedef float        v8f   __attribute__((ext_vector_type(8)));
typedef unsigned int u32x4 __attribute__((ext_vector_type(4)));
typedef int          i32x4 __attribute__((ext_vector_type(4)));
typedef int          i32x8 __attribute__((ext_vector_type(8)));

#if defined(__has_builtin)
#  if __has_builtin(__builtin_amdgcn_tensor_load_to_lds) && \
      __has_builtin(__builtin_amdgcn_s_wait_tensorcnt)
#    define HAVE_TDM 1
#  endif
#endif
#ifndef HAVE_TDM
#  define HAVE_TDM 0
#endif

// ---------------------------------------------------------------------------
// WMMA fragment helpers (16-bit operands, contiguous k = 16*(lane/16) + s, same order for A and B)
// ---------------------------------------------------------------------------
__device__ inline v16h load_contig16(const _Float16* __restrict__ p) {
  half8 lo = *reinterpret_cast<const half8*>(p);
  half8 hi = *reinterpret_cast<const half8*>(p + 8);
  v16h r;
#pragma unroll
  for (int i = 0; i < 8; ++i) { r[i] = lo[i]; r[8 + i] = hi[i]; }
  return r;
}

__device__ inline v16h load_a_frag(const _Float16* __restrict__ rowp, int lane) {
  const int kb = (lane & 16) ? 16 : 0;
  half8 lo = *reinterpret_cast<const half8*>(rowp + kb);
  half8 hi = *reinterpret_cast<const half8*>(rowp + kb + 8);
  v16h r;
#pragma unroll
  for (int i = 0; i < 8; ++i) { r[i] = lo[i]; r[8 + i] = hi[i]; }
  return r;
}

__device__ inline v16h load_b_frag(const _Float16* __restrict__ colp, int lane) {
  const int kb = (lane & 16) ? 16 : 0;
  return load_contig16(colp + kb);
}

// ---------------------------------------------------------------------------
// Generic 16-row-tile GEMM: C[M x 64] = A[M x Ktot] f16 x Bt[64 x Ktot] f16
// ---------------------------------------------------------------------------
__global__ __launch_bounds__(32)
void gemm_wmma_kernel(const _Float16* __restrict__ A, int lda,
                      const _Float16* __restrict__ Bt, int ldb,
                      float* __restrict__ C, int ldc, int Ktot) {
  const int lane = threadIdx.x;
  const int fcol = lane & 15;
  const size_t row = (size_t)blockIdx.x * 16 + fcol;
  v8f acc0 = {}, acc1 = {}, acc2 = {}, acc3 = {};
  for (int k0 = 0; k0 < Ktot; k0 += 32) {
    v16h a  = load_a_frag(A + row * lda + k0, lane);
    v16h b0 = load_b_frag(Bt + (size_t)(fcol)      * ldb + k0, lane);
    v16h b1 = load_b_frag(Bt + (size_t)(16 + fcol) * ldb + k0, lane);
    v16h b2 = load_b_frag(Bt + (size_t)(32 + fcol) * ldb + k0, lane);
    v16h b3 = load_b_frag(Bt + (size_t)(48 + fcol) * ldb + k0, lane);
    acc0 = __builtin_amdgcn_wmma_f32_16x16x32_f16(false, a, false, b0, (short)0, acc0, false, false);
    acc1 = __builtin_amdgcn_wmma_f32_16x16x32_f16(false, a, false, b1, (short)0, acc1, false, false);
    acc2 = __builtin_amdgcn_wmma_f32_16x16x32_f16(false, a, false, b2, (short)0, acc2, false, false);
    acc3 = __builtin_amdgcn_wmma_f32_16x16x32_f16(false, a, false, b3, (short)0, acc3, false, false);
  }
  const size_t mbase = (size_t)blockIdx.x * 16 + ((lane & 16) ? 8 : 0);
#pragma unroll
  for (int r = 0; r < 8; ++r) {
    float* cp = C + (mbase + r) * ldc + fcol;
    cp[0] = acc0[r]; cp[16] = acc1[r]; cp[32] = acc2[r]; cp[48] = acc3[r];
  }
}

// ---------------------------------------------------------------------------
// TDM: DMA one 32(j) x 64(f) f16 tile of h2t[64][N] into LDS, padded to an
// 80-byte row pitch via the TDM pad feature (pad_interval=16 DW, pad=4 DW)
// so ds_load_b128 fragment reads stay 16B-aligned and bank-conflict-free.
// D# layout per CDNA5 ISA.
// ---------------------------------------------------------------------------
#if HAVE_TDM
__device__ inline void tdm_load_tile(const _Float16* gbase, int j0, int N,
                                     unsigned lds_off) {
  unsigned long long ga =
      (unsigned long long)(size_t)gbase + (unsigned long long)j0 * 2ull;
  u32x4 g0;
  g0[0] = 1u;                                   // count=1 (valid user D#)
  g0[1] = lds_off;                              // lds_addr (bytes)
  g0[2] = (unsigned)(ga & 0xFFFFFFFFull);       // global_addr[31:0]
  g0[3] = (unsigned)((ga >> 32) & 0x1FFFFFFull) // global_addr[56:32]
          | (2u << 30);                         // type=2 ("image")
  i32x8 g1;
  g1[0] = (1 << 16)        // data_size = 2 bytes
        | (1 << 20)        // pad_enable
        | (3 << 22)        // pad_interval: 16 DWORDs (64B) between pads
        | (3 << 25);       // pad_amount:   4 DWORDs (16B)
  g1[1] = (int)(((unsigned)N & 0xFFFFu) << 16); // tensor_dim0[15:0]
  g1[2] = (HDIM << 16);                         // tensor_dim0 hi | tensor_dim1=64
  g1[3] = (32 << 16);                           // tile_dim0 = 32 elements
  g1[4] = HDIM;                                 // tile_dim1 = 64, tile_dim2 = 0
  g1[5] = N;                                    // tensor_dim0_stride (elements)
  g1[6] = 0;
  g1[7] = 0;
  i32x4 gz = {0, 0, 0, 0};
#if defined(__clang_major__) && (__clang_major__ >= 23)
  i32x8 gz8 = {0, 0, 0, 0, 0, 0, 0, 0};
  __builtin_amdgcn_tensor_load_to_lds(g0, g1, gz, gz, gz8, 0);
#else
  __builtin_amdgcn_tensor_load_to_lds(g0, g1, gz, gz, 0);
#endif
}
#endif

// ---------------------------------------------------------------------------
// Fused flash-style Gaussian-kernel aggregation with LDS-shared B tiles:
//   z[i,:] = (sum_j exp(-(s_j-s_i)^2) h2[j,:]) / sum_j exp(-(s_j-s_i)^2)
// 6 waves/block, each owns one 16-row i-tile; wave 0 double-buffers the
// shared 32x64 f16 B tile into LDS with TENSOR_LOAD_TO_LDS (TENSORcnt).
// kappa tile built in registers (shfl scores + v_exp_f32) as WMMA A operand;
// row sums via a 5th WMMA against a ones-column B fragment.
// ---------------------------------------------------------------------------
__global__ __launch_bounds__(KW * 32)
void kappa_z_wmma_kernel(const float* __restrict__ score,
                         const _Float16* __restrict__ h2t,  // [64][N]
                         float* __restrict__ z, int N) {
  __shared__ _Float16 tileS[2][HDIM * LDS_ROW];   // 2 x 5120B, 80B row pitch
  const int lane = threadIdx.x & 31;
  const int wave = threadIdx.x >> 5;
  const int tile = blockIdx.x * KW + wave;
  const int fcol = lane & 15;
  const float si = score[tile * 16 + fcol];
  const int ka = (lane & 16) ? 16 : 0;   // A-fragment K base
  const int kb = (lane & 16) ? 16 : 0;   // B-fragment K base (halfs)
  v8f accz0 = {}, accz1 = {}, accz2 = {}, accz3 = {}, accs = {};
  v16h bones;
  {
    _Float16 v = (fcol == 0) ? (_Float16)1.0f : (_Float16)0.0f;
#pragma unroll
    for (int i = 0; i < 16; ++i) bones[i] = v;
  }
  const int iters = N / 32;

#if HAVE_TDM
  if (wave == 0) {                       // TDM is per-wave; EXEC ignored
    tdm_load_tile(h2t, 0, N, (unsigned)(size_t)&tileS[0][0]);
    __builtin_amdgcn_s_wait_tensorcnt(0);
  }
  __syncthreads();
#endif

  for (int it = 0; it < iters; ++it) {
    const int cur = it & 1;
    const int j0 = it * 32;
#if HAVE_TDM
    if (wave == 0 && it + 1 < iters)     // kick next DMA into other buffer
      tdm_load_tile(h2t, j0 + 32, N, (unsigned)(size_t)&tileS[1 - cur][0]);
#else
    __syncthreads();                     // buffer reusable
    for (int t = threadIdx.x; t < HDIM * 4; t += KW * 32) {
      int r = t >> 2, c = (t & 3) * 8;   // 4 x half8 per 32-half row
      *reinterpret_cast<half8*>(&tileS[0][r * LDS_ROW + c]) =
          *reinterpret_cast<const half8*>(h2t + (size_t)r * N + j0 + c);
    }
    __syncthreads();
#endif
    // --- A operand: 16x32 kappa tile in registers ---
    float sjl = score[j0 + lane];        // one coalesced load per lane
    v16h a;
#pragma unroll
    for (int e = 0; e < 8; ++e) {        // K = ka + e
      float d = __shfl(sjl, ka + e, 32) - si;
      a[e] = (_Float16)__expf(-d * d);
    }
#pragma unroll
    for (int e = 0; e < 8; ++e) {        // K = ka + 8 + e
      float d = __shfl(sjl, ka + 8 + e, 32) - si;
      a[8 + e] = (_Float16)__expf(-d * d);
    }
    // --- B operands from LDS (ds_load_b128 x2 per fragment) ---
#if HAVE_TDM
    const _Float16* lb = &tileS[cur][0];
#else
    const _Float16* lb = &tileS[0][0];
#endif
    v16h b0 = load_contig16(lb + (size_t)(fcol)              * LDS_ROW + kb);
    v16h b1 = load_contig16(lb + (size_t)(16 + fcol)         * LDS_ROW + kb);
    v16h b2 = load_contig16(lb + (size_t)(32 + fcol)         * LDS_ROW + kb);
    v16h b3 = load_contig16(lb + (size_t)(48 + fcol)         * LDS_ROW + kb);
    accz0 = __builtin_amdgcn_wmma_f32_16x16x32_f16(false, a, false, b0, (short)0, accz0, false, false);
    accz1 = __builtin_amdgcn_wmma_f32_16x16x32_f16(false, a, false, b1, (short)0, accz1, false, false);
    accz2 = __builtin_amdgcn_wmma_f32_16x16x32_f16(false, a, false, b2, (short)0, accz2, false, false);
    accz3 = __builtin_amdgcn_wmma_f32_16x16x32_f16(false, a, false, b3, (short)0, accz3, false, false);
    accs  = __builtin_amdgcn_wmma_f32_16x16x32_f16(false, a, false, bones, (short)0, accs, false, false);
#if HAVE_TDM
    __syncthreads();                     // all waves done reading tileS[cur]
    if (wave == 0 && it + 1 < iters)
      __builtin_amdgcn_s_wait_tensorcnt(0);   // next buffer landed
    __syncthreads();                     // publish next buffer
#endif
  }

  const size_t mbase = (size_t)tile * 16 + ((lane & 16) ? 8 : 0);
#pragma unroll
  for (int r = 0; r < 8; ++r) {
    // rowsum for M=r lives in lane 0 (rows 0..7) / lane 16 (rows 8..15), col 0
    float lo = __shfl(accs[r], 0, 32);
    float hi = __shfl(accs[r], 16, 32);
    float inv = 1.0f / ((lane & 16) ? hi : lo);
    float* zp = z + (mbase + r) * HDIM + fcol;
    zp[0]  = accz0[r] * inv;
    zp[16] = accz1[r] * inv;
    zp[32] = accz2[r] * inv;
    zp[48] = accz3[r] * inv;
  }
}

// ------------------------- small support kernels ---------------------------
__global__ void set_f32_kernel(float* __restrict__ p, float v, int n) {
  int i = blockIdx.x * blockDim.x + threadIdx.x;
  if (i < n) p[i] = v;
}

__global__ void f32_to_f16_kernel(const float* __restrict__ s, _Float16* __restrict__ d, int n) {
  int i = blockIdx.x * blockDim.x + threadIdx.x;
  if (i < n) d[i] = (_Float16)s[i];
}

// dst[f*K + k] = (f16) src[k*F + f]
__global__ void transpose_f16_kernel(const float* __restrict__ src, _Float16* __restrict__ dst,
                                     int K, int F) {
  int i = blockIdx.x * blockDim.x + threadIdx.x;
  if (i >= K * F) return;
  int f = i / K, k = i % K;
  dst[i] = (_Float16)src[(size_t)k * F + f];
}

__global__ void count_deg_kernel(const int* __restrict__ ei, float* __restrict__ deg, int E) {
  int e = blockIdx.x * blockDim.x + threadIdx.x;
  if (e < E) atomicAdd(&deg[ei[E + e]], 1.0f);
}

__global__ void rsqrt_kernel(const float* __restrict__ deg, float* __restrict__ isq, int n) {
  int i = blockIdx.x * blockDim.x + threadIdx.x;
  if (i < n) isq[i] = rsqrtf(deg[i]);
}

__global__ __launch_bounds__(64)
void scatter_edges_kernel(const int* __restrict__ ei, const float* __restrict__ pre,
                          const float* __restrict__ isq, float* __restrict__ agg, int E) {
  int e = blockIdx.x;
  int f = threadIdx.x;
  int s = ei[e], d = ei[E + e];
  float nrm = isq[s] * isq[d];
  atomicAdd(&agg[(size_t)d * HDIM + f], pre[(size_t)s * HDIM + f] * nrm);
}

__global__ void finalize_kernel(const float* __restrict__ pre, const float* __restrict__ agg,
                                const float* __restrict__ deg, const float* __restrict__ bias,
                                float* __restrict__ of32, _Float16* __restrict__ of16,
                                int total, int relu) {
  int i = blockIdx.x * blockDim.x + threadIdx.x;
  if (i >= total) return;
  int node = i >> 6, f = i & 63;
  float v = agg[i] + pre[i] / deg[node] + bias[f];
  if (relu) v = fmaxf(v, 0.0f);
  if (of32) of32[i] = v;
  if (of16) of16[i] = (_Float16)v;
}

__global__ __launch_bounds__(32)
void auxnorm_kernel(const float* __restrict__ aux, float* __restrict__ a_n) {
  int lane = threadIdx.x;
  float v0 = aux[lane], v1 = aux[32 + lane];
  float s = v0 * v0 + v1 * v1;
#pragma unroll
  for (int off = 16; off > 0; off >>= 1) s += __shfl_xor(s, off, 32);
  s = __shfl(s, 0, 32);
  float nrm = fmaxf(sqrtf(s), 1e-12f);
  a_n[lane] = v0 / nrm;
  a_n[32 + lane] = v1 / nrm;
}

__global__ __launch_bounds__(32)
void score_kernel(const float* __restrict__ h2, const float* __restrict__ a_n,
                  float* __restrict__ score) {
  int n = blockIdx.x, lane = threadIdx.x;
  float v0 = h2[(size_t)n * HDIM + lane];
  float v1 = h2[(size_t)n * HDIM + 32 + lane];
  float d = v0 * a_n[lane] + v1 * a_n[32 + lane];
  float s2 = v0 * v0 + v1 * v1;
#pragma unroll
  for (int off = 16; off > 0; off >>= 1) {
    d += __shfl_xor(d, off, 32);
    s2 += __shfl_xor(s2, off, 32);
  }
  if (lane == 0) score[n] = d / fmaxf(sqrtf(s2), 1e-8f);
}

__global__ void classifier_kernel(const float* __restrict__ h2, const float* __restrict__ z,
                                  const float* __restrict__ W, const float* __restrict__ b,
                                  float* __restrict__ out, int total) {
  int i = blockIdx.x * blockDim.x + threadIdx.x;
  if (i >= total) return;
  int n = i / NCLS, c = i % NCLS;
  float s = b[c];
#pragma unroll 8
  for (int f = 0; f < HDIM; ++f) {
    s += h2[(size_t)n * HDIM + f] * W[f * NCLS + c];
    s += z[(size_t)n * HDIM + f] * W[(HDIM + f) * NCLS + c];
  }
  out[i] = s;
}

// ---------------------------------------------------------------------------
extern "C" void kernel_launch(void* const* d_in, const int* in_sizes, int n_in,
                              void* d_out, int out_size, void* d_ws, size_t ws_size,
                              hipStream_t stream) {
  (void)in_sizes; (void)n_in; (void)out_size; (void)ws_size;
  const float* x    = (const float*)d_in[0];
  const float* W1   = (const float*)d_in[1];
  const float* b1   = (const float*)d_in[2];
  const float* W2   = (const float*)d_in[3];
  const float* b2   = (const float*)d_in[4];
  const float* aux  = (const float*)d_in[5];
  const float* clfW = (const float*)d_in[6];
  const float* clfb = (const float*)d_in[7];
  const int*   ei   = (const int*)d_in[8];
  float* out = (float*)d_out;

  char* ws = (char*)d_ws;
  size_t off = 0;
  auto alloc = [&](size_t bytes) -> void* {
    void* p = ws + off;
    off = (off + bytes + 255) & ~(size_t)255;
    return p;
  };
  float*    deg   = (float*)alloc((size_t)N_NODES * 4);
  float*    isq   = (float*)alloc((size_t)N_NODES * 4);
  _Float16* xh    = (_Float16*)alloc((size_t)N_NODES * F_IN * 2);
  _Float16* W1t   = (_Float16*)alloc((size_t)HDIM * F_IN * 2);
  float*    h1pre = (float*)alloc((size_t)N_NODES * HDIM * 4);
  float*    agg   = (float*)alloc((size_t)N_NODES * HDIM * 4);
  _Float16* h1h   = (_Float16*)alloc((size_t)N_NODES * HDIM * 2);
  _Float16* W2t   = (_Float16*)alloc((size_t)HDIM * HDIM * 2);
  float*    h2pre = (float*)alloc((size_t)N_NODES * HDIM * 4);
  float*    h2    = (float*)alloc((size_t)N_NODES * HDIM * 4);
  _Float16* h2t   = (_Float16*)alloc((size_t)HDIM * N_NODES * 2);
  float*    a_n   = (float*)alloc((size_t)HDIM * 4);
  float*    score = (float*)alloc((size_t)N_NODES * 4);
  float*    zbuf  = (float*)alloc((size_t)N_NODES * HDIM * 4);

  const int TB = 256;
  const int NH = N_NODES * HDIM;

  // degree (with self loop) and D^{-1/2}
  set_f32_kernel<<<(N_NODES + TB - 1) / TB, TB, 0, stream>>>(deg, 1.0f, N_NODES);
  count_deg_kernel<<<(N_EDGES + TB - 1) / TB, TB, 0, stream>>>(ei, deg, N_EDGES);
  rsqrt_kernel<<<(N_NODES + TB - 1) / TB, TB, 0, stream>>>(deg, isq, N_NODES);

  // layer 1: h1pre = x @ W1 (WMMA f16, f32 accumulate)
  f32_to_f16_kernel<<<((N_NODES * F_IN) + TB - 1) / TB, TB, 0, stream>>>(x, xh, N_NODES * F_IN);
  transpose_f16_kernel<<<((HDIM * F_IN) + TB - 1) / TB, TB, 0, stream>>>(W1, W1t, F_IN, HDIM);
  gemm_wmma_kernel<<<N_NODES / 16, 32, 0, stream>>>(xh, F_IN, W1t, F_IN, h1pre, HDIM, F_IN);

  set_f32_kernel<<<(NH + TB - 1) / TB, TB, 0, stream>>>(agg, 0.0f, NH);
  scatter_edges_kernel<<<N_EDGES, 64, 0, stream>>>(ei, h1pre, isq, agg, N_EDGES);
  finalize_kernel<<<(NH + TB - 1) / TB, TB, 0, stream>>>(h1pre, agg, deg, b1,
                                                         nullptr, h1h, NH, /*relu=*/1);

  // layer 2: h2pre = relu(h1) @ W2
  transpose_f16_kernel<<<((HDIM * HDIM) + TB - 1) / TB, TB, 0, stream>>>(W2, W2t, HDIM, HDIM);
  gemm_wmma_kernel<<<N_NODES / 16, 32, 0, stream>>>(h1h, HDIM, W2t, HDIM, h2pre, HDIM, HDIM);

  set_f32_kernel<<<(NH + TB - 1) / TB, TB, 0, stream>>>(agg, 0.0f, NH);
  scatter_edges_kernel<<<N_EDGES, 64, 0, stream>>>(ei, h2pre, isq, agg, N_EDGES);
  finalize_kernel<<<(NH + TB - 1) / TB, TB, 0, stream>>>(h2pre, agg, deg, b2,
                                                         h2, nullptr, NH, /*relu=*/0);

  // transposed f16 copy of h2 for the kappa WMMA B-operand
  transpose_f16_kernel<<<((HDIM * N_NODES) + TB - 1) / TB, TB, 0, stream>>>(h2, h2t, N_NODES, HDIM);

  // cosine scores
  auxnorm_kernel<<<1, 32, 0, stream>>>(aux, a_n);
  score_kernel<<<N_NODES, 32, 0, stream>>>(h2, a_n, score);

  // fused N^2 Gaussian-kernel aggregation: 6 i-tiles/block share TDM-staged
  // LDS B tiles, double-buffered on TENSORcnt
  kappa_z_wmma_kernel<<<(N_NODES / 16) / KW, KW * 32, 0, stream>>>(score, h2t, zbuf, N_NODES);

  // classifier: concat([h2, z]) @ clf_W + clf_b
  classifier_kernel<<<((N_NODES * NCLS) + TB - 1) / TB, TB, 0, stream>>>(h2, zbuf, clfW, clfb,
                                                                         out, N_NODES * NCLS);
}
